// PatchcoreModel_7129645711947
// MI455X (gfx1250) — matrix operand order repaired and checked by reading, not compiled, not run on hardware
//
#include <hip/hip_runtime.h>
#include <hip/hip_bf16.h>
#include <math.h>

typedef __bf16 bf16_t;
typedef __attribute__((ext_vector_type(16))) __bf16 v16bf;
typedef __attribute__((ext_vector_type(8)))  __bf16 v8bf;
typedef __attribute__((ext_vector_type(8)))  float  v8f;

#define BATCH   16
#define C2      512
#define H2      28
#define W2      28
#define C3      1024
#define H3      14
#define W3      14
#define EMB     1536
#define PATCH   784                  // 28*28
#define NQ      (BATCH * PATCH)      // 12544
#define NCS     16384
#define IMG     224
#define KS      33
#define HALF_KS 16
#define SIGMA   4.0f

#define MT 128   // query rows per workgroup
#define NT 128   // coreset rows per N-step

// ---------------------------------------------------------------------------
// CDNA5 async global->LDS copy (ASYNCcnt-tracked, no VGPR round trip).
// LDS operand = low 32 bits of the generic shared pointer (flat-LDS aperture
// truncation, ISA 10.2). Copies 32 bytes (2 x b128) per call.
// ---------------------------------------------------------------------------
__device__ __forceinline__ void async_copy32(void* lds, const void* glob) {
  unsigned l0 = (unsigned)(size_t)lds;
  unsigned long long g0 = (unsigned long long)(size_t)glob;
  unsigned l1 = l0 + 16;
  unsigned long long g1 = g0 + 16ull;
  asm volatile("global_load_async_to_lds_b128 %0, %1, off"
               :: "v"(l0), "v"(g0) : "memory");
  asm volatile("global_load_async_to_lds_b128 %0, %1, off"
               :: "v"(l1), "v"(g1) : "memory");
}

__device__ __forceinline__ void wait_async0() {
  asm volatile("s_wait_asynccnt 0" ::: "memory");
}

// ---------------------------------------------------------------------------
// One K=32 stage: load fragments from an LDS stage buffer, run 8 WMMAs.
// Fragment layouts (16-bit A and B both contiguous-K halves, k = 16*(lane/16) + e;
// not the ISA table order). A sched_barrier separates the fragment-load phase
// from the WMMA phase so all 12 ds_load_b128 issue ahead of a back-to-back
// WMMA burst (keeps 4 distinct B-fragment register ranges live).
// ---------------------------------------------------------------------------
__device__ __forceinline__ void mma_stage(const bf16_t (&A)[MT][32],
                                          const bf16_t (&B)[NT][32],
                                          int wm, int wn, int lrow,
                                          int akb, int bkb,
                                          v8f (&acc)[2][4]) {
  v16bf afrag[2];
  #pragma unroll
  for (int s = 0; s < 2; ++s) {
    int row = wm * 32 + s * 16 + lrow;
    v8bf alo = *(const v8bf*)&A[row][akb];
    v8bf ahi = *(const v8bf*)&A[row][akb + 8];
    #pragma unroll
    for (int e = 0; e < 8; ++e) { afrag[s][e] = alo[e]; afrag[s][e + 8] = ahi[e]; }
  }
  v16bf bfrag[4];
  #pragma unroll
  for (int c = 0; c < 4; ++c) {
    int nrow = wn * 64 + c * 16 + lrow;
    v8bf blo = *(const v8bf*)&B[nrow][bkb];
    v8bf bhi = *(const v8bf*)&B[nrow][bkb + 8];
    #pragma unroll
    for (int e = 0; e < 8; ++e) { bfrag[c][e] = blo[e]; bfrag[c][e + 8] = bhi[e]; }
  }
  __builtin_amdgcn_sched_barrier(0);   // loads stay above, WMMAs below
  #pragma unroll
  for (int c = 0; c < 4; ++c) {
    acc[0][c] = __builtin_amdgcn_wmma_f32_16x16x32_bf16(
        false, afrag[0], false, bfrag[c], (short)0, acc[0][c], false, false);
    acc[1][c] = __builtin_amdgcn_wmma_f32_16x16x32_bf16(
        false, afrag[1], false, bfrag[c], (short)0, acc[1][c], false, false);
  }
}

// ---------------------------------------------------------------------------
// Fused AvgPool2d(3,1,1, count_include_pad) + nearest-2x-upsample + concat.
// ---------------------------------------------------------------------------
__device__ __forceinline__ float embed_value(const float* __restrict__ f2,
                                             const float* __restrict__ f3,
                                             int b, int i, int j, int c) {
  float s = 0.0f;
  if (c < C2) {
    const float* base = f2 + ((size_t)(b * C2 + c)) * (H2 * W2);
    #pragma unroll
    for (int di = -1; di <= 1; ++di) {
      int ii = i + di;
      if (ii < 0 || ii >= H2) continue;
      #pragma unroll
      for (int dj = -1; dj <= 1; ++dj) {
        int jj = j + dj;
        if (jj < 0 || jj >= W2) continue;
        s += base[ii * W2 + jj];
      }
    }
  } else {
    int i3 = i >> 1, j3 = j >> 1;
    const float* base = f3 + ((size_t)(b * C3 + (c - C2))) * (H3 * W3);
    #pragma unroll
    for (int di = -1; di <= 1; ++di) {
      int ii = i3 + di;
      if (ii < 0 || ii >= H3) continue;
      #pragma unroll
      for (int dj = -1; dj <= 1; ++dj) {
        int jj = j3 + dj;
        if (jj < 0 || jj >= W3) continue;
        s += base[ii * W3 + jj];
      }
    }
  }
  return s * (1.0f / 9.0f);
}

// ---------------------------------------------------------------------------
// 1) Build bf16 query matrix x[NQ][EMB] + f32 row norms (of bf16 values).
// ---------------------------------------------------------------------------
__global__ __launch_bounds__(256) void embed_kernel(const float* __restrict__ f2,
                                                    const float* __restrict__ f3,
                                                    bf16_t* __restrict__ xb,
                                                    float* __restrict__ xn) {
  int q = blockIdx.x;
  int b = q / PATCH;
  int rem = q - b * PATCH;
  int i = rem / W2;
  int j = rem - i * W2;
  float acc = 0.0f;
  for (int c = threadIdx.x; c < EMB; c += 256) {
    float v = embed_value(f2, f3, b, i, j, c);
    bf16_t h = (bf16_t)v;
    xb[(size_t)q * EMB + c] = h;
    float fv = (float)h;
    acc = fmaf(fv, fv, acc);
  }
  __shared__ float red[256];
  red[threadIdx.x] = acc;
  __syncthreads();
  for (int s = 128; s > 0; s >>= 1) {
    if (threadIdx.x < s) red[threadIdx.x] += red[threadIdx.x + s];
    __syncthreads();
  }
  if (threadIdx.x == 0) xn[q] = red[0];
}

// ---------------------------------------------------------------------------
// 2) Coreset f32 -> bf16 + f32 row norms (of bf16 values).
// ---------------------------------------------------------------------------
__global__ __launch_bounds__(256) void coreset_prep_kernel(const float* __restrict__ cs,
                                                           bf16_t* __restrict__ cb,
                                                           float* __restrict__ cn) {
  int n = blockIdx.x;
  const float* src = cs + (size_t)n * EMB;
  float acc = 0.0f;
  for (int c = threadIdx.x; c < EMB; c += 256) {
    bf16_t h = (bf16_t)src[c];
    cb[(size_t)n * EMB + c] = h;
    float fv = (float)h;
    acc = fmaf(fv, fv, acc);
  }
  __shared__ float red[256];
  red[threadIdx.x] = acc;
  __syncthreads();
  for (int s = 128; s > 0; s >>= 1) {
    if (threadIdx.x < s) red[threadIdx.x] += red[threadIdx.x + s];
    __syncthreads();
  }
  if (threadIdx.x == 0) cn[n] = red[0];
}

// ---------------------------------------------------------------------------
// 3) WMMA GEMM (x . coreset^T) fused with min/argmin epilogue.
//    2-stage async pipeline, manually unrolled 2 K-stages per iteration:
//    each loop body = 2 x (wait_asynccnt + barrier + async-prefetch + 8 WMMA).
//    min key = 0.5*|c|^2 - dot  (|x|^2 added once at the end).
// ---------------------------------------------------------------------------
__global__ __launch_bounds__(256) void nn_kernel(const bf16_t* __restrict__ xb,
                                                 const bf16_t* __restrict__ cb,
                                                 const float* __restrict__ xn,
                                                 const float* __restrict__ cn,
                                                 float* __restrict__ pscore,
                                                 int* __restrict__ ploc) {
  __shared__ bf16_t sA[2][MT][32];
  __shared__ bf16_t sB[2][NT][32];
  __shared__ float  sxn[MT];
  __shared__ float  scn[NT];
  __shared__ float  pmin[2][MT];
  __shared__ int    pidx[2][MT];

  const int tid  = threadIdx.x;
  const int lane = tid & 31;
  const int wave = tid >> 5;
  const int wm   = wave & 3;
  const int wn   = wave >> 2;
  const int qbase = blockIdx.x * MT;

  if (tid < MT) sxn[tid] = xn[qbase + tid];

  const int lrow = lane & 15;
  const int akb  = (lane >> 4) * 16;   // A frag K-base, contiguous-K halves (matches B)
  const int bkb  = (lane >> 4) * 16;   // B frag K-base, contiguous-K halves (matches A)
  const int ldrow = tid >> 1;          // staging: 2 threads per row, 32B each
  const int ldcol = (tid & 1) * 16;

  const bf16_t* agp = xb + (size_t)(qbase + ldrow) * EMB + ldcol;

  float runmin[16];
  int   runidx[16];
  #pragma unroll
  for (int r = 0; r < 16; ++r) { runmin[r] = 3.402823e38f; runidx[r] = 0; }

  for (int nt = 0; nt < NCS; nt += NT) {
    __syncthreads();                    // previous epilogue/LDS reads done
    if (tid < NT) scn[tid] = cn[nt + tid];
    const bf16_t* bgp = cb + (size_t)(nt + ldrow) * EMB + ldcol;

    // prime stage buffer 0 with K chunk 0
    async_copy32(&sA[0][ldrow][ldcol], agp);
    async_copy32(&sB[0][ldrow][ldcol], bgp);

    v8f acc[2][4];
    #pragma unroll
    for (int s = 0; s < 2; ++s)
      #pragma unroll
      for (int c = 0; c < 4; ++c) {
        v8f z = {0.f, 0.f, 0.f, 0.f, 0.f, 0.f, 0.f, 0.f};
        acc[s][c] = z;
      }

    for (int kt = 0; kt < EMB; kt += 64) {
      // ---- stage A: compute buf0 (K chunk kt), prefetch buf1 (kt+32) ----
      wait_async0();
      __syncthreads();
      {
        int nxt = kt + 32;              // always < EMB
        async_copy32(&sA[1][ldrow][ldcol], agp + nxt);
        async_copy32(&sB[1][ldrow][ldcol], bgp + nxt);
      }
      mma_stage(sA[0], sB[0], wm, wn, lrow, akb, bkb, acc);

      // ---- stage B: compute buf1 (K chunk kt+32), prefetch buf0 (kt+64) --
      wait_async0();
      __syncthreads();
      if (kt + 64 < EMB) {
        int nxt = kt + 64;
        async_copy32(&sA[0][ldrow][ldcol], agp + nxt);
        async_copy32(&sB[0][ldrow][ldcol], bgp + nxt);
      }
      mma_stage(sA[1], sB[1], wm, wn, lrow, akb, bkb, acc);
    }

    // epilogue: key = 0.5*|c|^2 - dot ; fold into running min/argmin
    const int mofs = (lane >> 4) * 8;
    #pragma unroll
    for (int s = 0; s < 2; ++s)
      #pragma unroll
      for (int c = 0; c < 4; ++c) {
        int ncol_l = wn * 64 + c * 16 + lrow;
        int nglob  = nt + ncol_l;
        float cnh  = 0.5f * scn[ncol_l];
        #pragma unroll
        for (int r = 0; r < 8; ++r) {
          float key = cnh - acc[s][c][r];
          int slot = s * 8 + r;
          if (key < runmin[slot]) { runmin[slot] = key; runidx[slot] = nglob; }
        }
      }
  }

  // cross-lane reduce within 16-lane halves (wave32 shfl_xor)
  #pragma unroll
  for (int slot = 0; slot < 16; ++slot) {
    float m = runmin[slot];
    int   ix = runidx[slot];
    #pragma unroll
    for (int off = 1; off < 16; off <<= 1) {
      float m2 = __shfl_xor(m, off, 32);
      int   i2 = __shfl_xor(ix, off, 32);
      if (m2 < m || (m2 == m && i2 < ix)) { m = m2; ix = i2; }
    }
    runmin[slot] = m;
    runidx[slot] = ix;
  }
  if ((lane & 15) == 0) {
    int half = lane >> 4;
    #pragma unroll
    for (int slot = 0; slot < 16; ++slot) {
      int s = slot >> 3, r = slot & 7;
      int m_local = wm * 32 + s * 16 + r + half * 8;
      pmin[wn][m_local] = runmin[slot];
      pidx[wn][m_local] = runidx[slot];
    }
  }
  __syncthreads();
  if (tid < MT) {
    float m0 = pmin[0][tid], m1 = pmin[1][tid];
    int   i0 = pidx[0][tid], i1 = pidx[1][tid];
    float m; int ix;
    if (m1 < m0 || (m1 == m0 && i1 < i0)) { m = m1; ix = i1; } else { m = m0; ix = i0; }
    int q = qbase + tid;
    float d2 = sxn[tid] + 2.0f * m;
    pscore[q] = sqrtf(fmaxf(d2, 1e-12f));
    ploc[q] = ix;
  }
}

// ---------------------------------------------------------------------------
// 4) Per-batch argmax over patch scores (first-occurrence semantics).
// ---------------------------------------------------------------------------
__global__ __launch_bounds__(256) void batch_max_kernel(const float* __restrict__ ps,
                                                        const int* __restrict__ loc,
                                                        float* __restrict__ bscore,
                                                        int* __restrict__ bq,
                                                        int* __restrict__ bnn) {
  int b = blockIdx.x, tid = threadIdx.x;
  __shared__ float sm[256];
  __shared__ int   si[256];
  float best = -3.402823e38f;
  int   bi = 0x7fffffff;
  for (int p = tid; p < PATCH; p += 256) {
    float v = ps[b * PATCH + p];
    if (v > best || (v == best && p < bi)) { best = v; bi = p; }
  }
  sm[tid] = best; si[tid] = bi;
  __syncthreads();
  for (int s = 128; s > 0; s >>= 1) {
    if (tid < s) {
      if (sm[tid + s] > sm[tid] || (sm[tid + s] == sm[tid] && si[tid + s] < si[tid])) {
        sm[tid] = sm[tid + s]; si[tid] = si[tid + s];
      }
    }
    __syncthreads();
  }
  if (tid == 0) {
    bscore[b] = sm[0];
    bq[b] = si[0];
    bnn[b] = loc[b * PATCH + si[0]];
  }
}

// ---------------------------------------------------------------------------
// 5) Top-9 reweighting (exact f32 path; tiny vs the main GEMM).
// ---------------------------------------------------------------------------
__global__ __launch_bounds__(256) void score_kernel(const float* __restrict__ cs,
                                                    const float* __restrict__ f2,
                                                    const float* __restrict__ f3,
                                                    const float* __restrict__ bscore,
                                                    const int* __restrict__ bq,
                                                    const int* __restrict__ bnn,
                                                    float* __restrict__ out_scores) {
  __shared__ float nnf[EMB];
  __shared__ float mpf[EMB];
  __shared__ float tv_all[256 * 9];
  __shared__ int   ti_all[256 * 9];
  __shared__ float red[256];
  __shared__ int   sup[9];
  __shared__ float dsup[9];

  int b = blockIdx.x, tid = threadIdx.x;
  int nn = bnn[b];
  int qp = bq[b];
  int i = qp / W2, j = qp - i * W2;
  for (int c = tid; c < EMB; c += 256) {
    nnf[c] = cs[(size_t)nn * EMB + c];
    mpf[c] = embed_value(f2, f3, b, i, j, c);
  }
  __syncthreads();

  float tv[9]; int ti[9];
  #pragma unroll
  for (int k = 0; k < 9; ++k) { tv[k] = 3.402823e38f; ti[k] = 0x7fffffff; }
  for (int n = tid; n < NCS; n += 256) {
    const float* row = cs + (size_t)n * EMB;
    float d2 = 0.0f;
    for (int c = 0; c < EMB; ++c) {
      float df = row[c] - nnf[c];
      d2 = fmaf(df, df, d2);
    }
    float d = sqrtf(fmaxf(d2, 1e-12f));
    if (d < tv[8] || (d == tv[8] && n < ti[8])) {
      int k = 8;
      while (k > 0 && (d < tv[k - 1] || (d == tv[k - 1] && n < ti[k - 1]))) {
        tv[k] = tv[k - 1]; ti[k] = ti[k - 1]; --k;
      }
      tv[k] = d; ti[k] = n;
    }
  }
  #pragma unroll
  for (int k = 0; k < 9; ++k) { tv_all[tid * 9 + k] = tv[k]; ti_all[tid * 9 + k] = ti[k]; }
  __syncthreads();

  if (tid == 0) {
    float bv[9]; int bi9[9];
    #pragma unroll
    for (int k = 0; k < 9; ++k) { bv[k] = 3.402823e38f; bi9[k] = 0x7fffffff; }
    for (int t = 0; t < 256 * 9; ++t) {
      float d = tv_all[t];
      int   n = ti_all[t];
      if (d < bv[8] || (d == bv[8] && n < bi9[8])) {
        int k = 8;
        while (k > 0 && (d < bv[k - 1] || (d == bv[k - 1] && n < bi9[k - 1]))) {
          bv[k] = bv[k - 1]; bi9[k] = bi9[k - 1]; --k;
        }
        bv[k] = d; bi9[k] = n;
      }
    }
    #pragma unroll
    for (int k = 0; k < 9; ++k) sup[k] = bi9[k];
  }
  __syncthreads();

  for (int k = 0; k < 9; ++k) {
    const float* srow = cs + (size_t)sup[k] * EMB;
    float part = 0.0f;
    for (int c = tid; c < EMB; c += 256) {
      float df = mpf[c] - srow[c];
      part = fmaf(df, df, part);
    }
    red[tid] = part;
    __syncthreads();
    for (int s = 128; s > 0; s >>= 1) {
      if (tid < s) red[tid] += red[tid + s];
      __syncthreads();
    }
    if (tid == 0) dsup[k] = sqrtf(fmaxf(red[0], 1e-12f));
    __syncthreads();
  }

  if (tid == 0) {
    float mx = dsup[0];
    #pragma unroll
    for (int k = 1; k < 9; ++k) mx = fmaxf(mx, dsup[k]);
    float sum = 0.0f, e0 = 0.0f;
    #pragma unroll
    for (int k = 0; k < 9; ++k) {
      float e = expf(dsup[k] - mx);
      sum += e;
      if (k == 0) e0 = e;
    }
    float w = 1.0f - e0 / sum;
    out_scores[b] = w * bscore[b];
  }
}

// ---------------------------------------------------------------------------
// 6) Bilinear resize 28x28 -> 224x224 (half-pixel centers, edge clamp).
// ---------------------------------------------------------------------------
__global__ __launch_bounds__(256) void resize_kernel(const float* __restrict__ ps,
                                                     float* __restrict__ rmap) {
  int idx = blockIdx.x * 256 + threadIdx.x;
  if (idx >= BATCH * IMG * IMG) return;
  int b = idx / (IMG * IMG);
  int rem = idx - b * IMG * IMG;
  int y = rem / IMG;
  int x = rem - y * IMG;
  float fy = (y + 0.5f) * 0.125f - 0.5f;
  float fx = (x + 0.5f) * 0.125f - 0.5f;
  int y0 = (int)floorf(fy); float ty = fy - (float)y0;
  int x0 = (int)floorf(fx); float tx = fx - (float)x0;
  int y0c = min(max(y0, 0), 27), y1c = min(max(y0 + 1, 0), 27);
  int x0c = min(max(x0, 0), 27), x1c = min(max(x0 + 1, 0), 27);
  const float* p = ps + b * PATCH;
  float v00 = p[y0c * 28 + x0c], v01 = p[y0c * 28 + x1c];
  float v10 = p[y1c * 28 + x0c], v11 = p[y1c * 28 + x1c];
  float v = (1.0f - ty) * ((1.0f - tx) * v00 + tx * v01) +
            ty * ((1.0f - tx) * v10 + tx * v11);
  rmap[idx] = v;
}

// ---------------------------------------------------------------------------
// 7) Separable Gaussian blur (33 taps, sigma=4, reflect border).
// ---------------------------------------------------------------------------
__device__ __forceinline__ int reflect_idx(int r) {
  if (r < 0) r = -r;
  if (r >= IMG) r = 2 * IMG - 2 - r;
  return r;
}

__device__ __forceinline__ void make_gauss(float* wk, int tid) {
  if (tid == 0) {
    float s = 0.0f;
    for (int t = 0; t < KS; ++t) {
      float d = (float)(t - HALF_KS);
      wk[t] = expf(-(d * d) / (2.0f * SIGMA * SIGMA));
      s += wk[t];
    }
    float inv = 1.0f / s;
    for (int t = 0; t < KS; ++t) wk[t] *= inv;
  }
}

__global__ __launch_bounds__(256) void blur_h_kernel(const float* __restrict__ in,
                                                     float* __restrict__ out) {
  __shared__ float wk[KS];
  make_gauss(wk, threadIdx.x);
  __syncthreads();
  int idx = blockIdx.x * 256 + threadIdx.x;
  if (idx >= BATCH * IMG * IMG) return;
  int b = idx / (IMG * IMG);
  int rem = idx - b * IMG * IMG;
  int y = rem / IMG;
  int x = rem - y * IMG;
  const float* row = in + (size_t)(b * IMG + y) * IMG;
  float acc = 0.0f;
  #pragma unroll
  for (int t = 0; t < KS; ++t) acc = fmaf(wk[t], row[reflect_idx(x + t - HALF_KS)], acc);
  out[idx] = acc;
}

__global__ __launch_bounds__(256) void blur_v_kernel(const float* __restrict__ in,
                                                     float* __restrict__ out) {
  __shared__ float wk[KS];
  make_gauss(wk, threadIdx.x);
  __syncthreads();
  int idx = blockIdx.x * 256 + threadIdx.x;
  if (idx >= BATCH * IMG * IMG) return;
  int b = idx / (IMG * IMG);
  int rem = idx - b * IMG * IMG;
  int y = rem / IMG;
  int x = rem - y * IMG;
  const float* img = in + (size_t)b * IMG * IMG;
  float acc = 0.0f;
  #pragma unroll
  for (int t = 0; t < KS; ++t) acc = fmaf(wk[t], img[reflect_idx(y + t - HALF_KS) * IMG + x], acc);
  out[idx] = acc;
}

// ---------------------------------------------------------------------------
// Launch
// ---------------------------------------------------------------------------
extern "C" void kernel_launch(void* const* d_in, const int* in_sizes, int n_in,
                              void* d_out, int out_size, void* d_ws, size_t ws_size,
                              hipStream_t stream) {
  (void)in_sizes; (void)n_in; (void)out_size; (void)ws_size;
  const float* f2 = (const float*)d_in[0];   // [16,512,28,28]
  const float* f3 = (const float*)d_in[1];   // [16,1024,14,14]
  const float* cs = (const float*)d_in[2];   // [16384,1536]

  char* ws = (char*)d_ws;
  size_t off = 0;
  auto alloc = [&](size_t bytes) -> char* {
    char* p = ws + off;
    off += (bytes + 255) & ~(size_t)255;
    return p;
  };
  bf16_t* xb     = (bf16_t*)alloc((size_t)NQ * EMB * sizeof(bf16_t));
  bf16_t* cb     = (bf16_t*)alloc((size_t)NCS * EMB * sizeof(bf16_t));
  float*  xn     = (float*)alloc((size_t)NQ * sizeof(float));
  float*  cn     = (float*)alloc((size_t)NCS * sizeof(float));
  float*  pscore = (float*)alloc((size_t)NQ * sizeof(float));
  int*    ploc   = (int*)alloc((size_t)NQ * sizeof(int));
  float*  bscore = (float*)alloc(BATCH * sizeof(float));
  int*    bq     = (int*)alloc(BATCH * sizeof(int));
  int*    bnn    = (int*)alloc(BATCH * sizeof(int));
  float*  rmap   = (float*)alloc((size_t)BATCH * IMG * IMG * sizeof(float));
  float*  tmap   = (float*)alloc((size_t)BATCH * IMG * IMG * sizeof(float));

  float* out_map    = (float*)d_out;                       // [16,1,224,224]
  float* out_scores = out_map + (size_t)BATCH * IMG * IMG; // [16]

  embed_kernel<<<NQ, 256, 0, stream>>>(f2, f3, xb, xn);
  coreset_prep_kernel<<<NCS, 256, 0, stream>>>(cs, cb, cn);
  nn_kernel<<<NQ / MT, 256, 0, stream>>>(xb, cb, xn, cn, pscore, ploc);
  batch_max_kernel<<<BATCH, 256, 0, stream>>>(pscore, ploc, bscore, bq, bnn);
  score_kernel<<<BATCH, 256, 0, stream>>>(cs, f2, f3, bscore, bq, bnn, out_scores);

  int npix = BATCH * IMG * IMG;
  resize_kernel<<<(npix + 255) / 256, 256, 0, stream>>>(pscore, rmap);
  blur_h_kernel<<<(npix + 255) / 256, 256, 0, stream>>>(rmap, tmap);
  blur_v_kernel<<<(npix + 255) / 256, 256, 0, stream>>>(tmap, out_map);
}
